// ABGCN_51092930953449
// MI455X (gfx1250) — compile-verified, matrix operand order checked by reading, not run on hardware
//
#include <hip/hip_runtime.h>
#include <hip/hip_bf16.h>

typedef __attribute__((ext_vector_type(16))) __bf16 v16bf;
typedef __attribute__((ext_vector_type(8)))  float  v8f;
typedef __attribute__((ext_vector_type(4)))  unsigned int u32x4;
typedef __attribute__((ext_vector_type(8)))  unsigned int u32x8;

#define SEQL   64
#define DMODEL 300
#define KPAD   320
#define NKV    600
#define NKVPAD 608
#define NHEAD  5
#define HD     60
#define S2VD   256
#define VOCABN 50001

// CDNA5 16-bit A/B operand K-index for vector element e, lane-group g
// (ISA 7.12.2: VGPR0..3 hold K pairs 0..7 (g=0) / 8..15 (g=1), VGPR4..7 hold +16)
__device__ __forceinline__ int wmma_k(int e, int g) {
    int v = e >> 1;
    return ((v >> 2) << 4) + (g << 3) + ((v & 3) << 1) + (e & 1);
}

// ---- weight convert: out[R][Cpad] bf16 from W[(row0+r)*C + c], zero-padded
__global__ void k_cvt_bf16(const float* __restrict__ W, __bf16* __restrict__ out,
                           int R, int Rin, int Cpad, int C, int row0) {
    int total = R * Cpad;
    for (int idx = blockIdx.x * blockDim.x + threadIdx.x; idx < total;
         idx += gridDim.x * blockDim.x) {
        int row = idx / Cpad, col = idx - row * Cpad;
        float v = (row < Rin && col < C) ? W[(size_t)(row0 + row) * C + col] : 0.f;
        out[idx] = (__bf16)v;
    }
}

__global__ void k_f32_to_bf16(const float* __restrict__ in, __bf16* __restrict__ out, int n) {
    for (int i = blockIdx.x * blockDim.x + threadIdx.x; i < n; i += gridDim.x * blockDim.x)
        out[i] = (__bf16)in[i];
}

__global__ void k_zero(float* p, int n) {
    for (int i = blockIdx.x * blockDim.x + threadIdx.x; i < n; i += gridDim.x * blockDim.x)
        p[i] = 0.f;
}

// ---- fused per-post: TDM-gather embeddings -> bf16 WMMA KV gemm ->
//      1-query attention -> out-proj -> tanh(s2v).  One block (8 wave32) per post.
__global__ __launch_bounds__(256) void k_post_encoder(
    const int*   __restrict__ tokens,   // [NPOSTS][64]
    const float* __restrict__ embed,    // [VOCAB][300]
    const float* __restrict__ inW,      // [900][300]
    const float* __restrict__ inB,      // [900]
    const __bf16* __restrict__ wkv,     // [608][320] bf16 (rows = qkv col 300+n)
    const float* __restrict__ outW,     // [300][300]
    const float* __restrict__ outB,     // [300]
    const float* __restrict__ s2vW,     // [256][300]
    const float* __restrict__ s2vB,     // [256]
    float*       __restrict__ s2v)      // [NPOSTS][256]
{
    __shared__ __bf16 Xs[SEQL][KPAD];        // 40 KB
    __shared__ __bf16 KVs[SEQL][NKVPAD];     // 76 KB: cols 0..299 = K, 300..599 = V
    __shared__ float  stage[16 * DMODEL];    // 18.75 KB TDM gather landing buffer
    __shared__ float  x0f[DMODEL];           // f32 copy of token-0 embedding
    __shared__ float  biasKV[NKVPAD];
    __shared__ float  q0s[DMODEL];
    __shared__ float  att[NHEAD][SEQL];
    __shared__ float  o0s[DMODEL];
    __shared__ float  ys[DMODEL];

    const int post = blockIdx.x;
    const int tid  = threadIdx.x;
    const int wid  = tid >> 5, lane = tid & 31;
    const int r = lane & 15, g = lane >> 4;

    // zero-padded K/V bias slice (avoids divergent guard at WMMA writeback)
    for (int n = tid; n < NKVPAD; n += 256)
        biasKV[n] = (n < NKV) ? inB[DMODEL + n] : 0.f;

    // Stage A: gather 64 embedding rows via TDM gather-mode tensor loads.
    // D# per CDNA5 ISA: gather_mode=1,
    // 16-bit row indices (VOCAB=50001 < 2^16), 16 rows per descriptor,
    // row width 300 f32, packed contiguously at lds_addr.
    const unsigned lds_base = (unsigned)(size_t)(void*)stage;
    const unsigned long long ga = (unsigned long long)(size_t)embed;
    for (int c = 0; c < 4; ++c) {
        if (wid == 0) {
            const int base = c * 16;
            u32x4 g0;
            g0[0] = 1u | (1u << 31);                          // count=1, gather_mode=1, idx16
            g0[1] = lds_base;                                  // lds_addr
            g0[2] = (unsigned)(ga & 0xFFFFFFFFu);              // global_addr[31:0]
            g0[3] = (unsigned)((ga >> 32) & 0x01FFFFFFu) | (2u << 30); // addr[56:32] | type=2
            u32x8 g1;
            g1[0] = (2u << 16);                                // data_size=4B
            g1[1] = ((unsigned)DMODEL & 0xFFFFu) << 16;        // tensor_dim0[15:0]
            g1[2] = ((unsigned)VOCABN & 0xFFFFu) << 16;        // tensor_dim1[15:0]
            g1[3] = (((unsigned)VOCABN >> 16) & 0xFFFFu) |
                    ((unsigned)DMODEL << 16);                  // tensor_dim1[31:16] | tile_dim0
            g1[4] = 16u;                                       // tile_dim1 = #valid indices
            g1[5] = (unsigned)DMODEL;                          // tensor_dim0_stride[31:0]
            g1[6] = 0u;
            g1[7] = 0u;
            unsigned idx[8];
#pragma unroll
            for (int j = 0; j < 8; ++j) {
                unsigned a = (unsigned)tokens[post * SEQL + base + 2 * j];
                unsigned b = (unsigned)tokens[post * SEQL + base + 2 * j + 1];
                idx[j] = (a & 0xFFFFu) | (b << 16);
            }
            u32x4 g2 = {idx[0], idx[1], idx[2], idx[3]};
            u32x4 g3 = {idx[4], idx[5], idx[6], idx[7]};
            asm volatile("tensor_load_to_lds %0, %1, %2, %3"
                         :: "s"(g0), "s"(g1), "s"(g2), "s"(g3)
                         : "memory");
            __builtin_amdgcn_s_wait_tensorcnt(0);
        }
        __syncthreads();
        // convert staged f32 rows -> bf16 A-matrix (K padded to 320)
        for (int idx2 = tid; idx2 < 16 * KPAD; idx2 += 256) {
            int t = idx2 / KPAD, k = idx2 - t * KPAD;
            float v = (k < DMODEL) ? stage[t * DMODEL + k] : 0.f;
            Xs[c * 16 + t][k] = (__bf16)v;
            if (c == 0 && t == 0 && k < DMODEL) x0f[k] = v;
        }
        __syncthreads();
    }

    // Stage B: KV = X(64x320) @ Wkv^T(320x608) via v_wmma_f32_16x16x32_bf16
    for (int tt = wid; tt < 4 * (NKVPAD / 16); tt += 8) {
        int mt = tt & 3, nt = tt >> 2;
        const __bf16* brow = wkv + (size_t)(nt * 16 + r) * KPAD;
        v8f acc = {};
        for (int kc = 0; kc < KPAD / 32; ++kc) {
            int kb = kc * 32;
            v16bf a, b;
#pragma unroll
            for (int e = 0; e < 16; ++e) {
                int kk = kb + wmma_k(e, g);
                a[e] = Xs[mt * 16 + r][kk];
                b[e] = brow[kk];
            }
            acc = __builtin_amdgcn_wmma_f32_16x16x32_bf16(
                false, a, false, b, (short)0, acc, false, false);
        }
#pragma unroll
        for (int i = 0; i < 8; ++i) {
            int m = mt * 16 + g * 8 + i;
            int n = nt * 16 + r;
            KVs[m][n] = (__bf16)(acc[i] + biasKV[n]);
        }
    }
    __syncthreads();

    // Stage C: q0 = X[0] @ Wq^T + bq  (only query position 0 is consumed)
    for (int j = tid; j < DMODEL; j += 256) {
        float s = inB[j];
        for (int k = 0; k < DMODEL; ++k) s += x0f[k] * inW[(size_t)j * DMODEL + k];
        q0s[j] = s;
    }
    __syncthreads();

    // Stage D: scores[h][k] = (q0_h . K_k,h) / sqrt(60)
    for (int i = tid; i < NHEAD * SEQL; i += 256) {
        int h = i / SEQL, kk = i - h * SEQL;
        float s = 0.f;
        for (int d = 0; d < HD; ++d) s += q0s[h * HD + d] * (float)KVs[kk][h * HD + d];
        att[h][kk] = s * 0.12909944487358056f;
    }
    __syncthreads();
    if (tid < NHEAD) {
        float m = -1e30f;
        for (int k = 0; k < SEQL; ++k) m = fmaxf(m, att[tid][k]);
        float sum = 0.f;
        for (int k = 0; k < SEQL; ++k) { float e = __expf(att[tid][k] - m); att[tid][k] = e; sum += e; }
        float inv = 1.f / sum;
        for (int k = 0; k < SEQL; ++k) att[tid][k] *= inv;
    }
    __syncthreads();

    // Stage E: o0 = att @ V
    for (int j = tid; j < DMODEL; j += 256) {
        int h = j / HD, d = j - h * HD;
        float s = 0.f;
        for (int k = 0; k < SEQL; ++k) s += att[h][k] * (float)KVs[k][300 + h * HD + d];
        o0s[j] = s;
    }
    __syncthreads();

    // Stage F: y = o0 @ Wo^T + bo
    for (int i = tid; i < DMODEL; i += 256) {
        float s = outB[i];
        for (int j = 0; j < DMODEL; ++j) s += o0s[j] * outW[(size_t)i * DMODEL + j];
        ys[i] = s;
    }
    __syncthreads();

    // Stage G: s2v = tanh(y @ Ws^T + bs)
    for (int c = tid; c < S2VD; c += 256) {
        float s = s2vB[c];
        for (int i = 0; i < DMODEL; ++i) s += ys[i] * s2vW[(size_t)c * DMODEL + i];
        s2v[(size_t)post * S2VD + c] = tanhf(s);
    }
}

// ---- generic C[M][N] = A[M][K]bf16 @ B[N][K]bf16^T, wave-per-tile WMMA
__global__ __launch_bounds__(256) void k_gemm_bf16(
    const __bf16* __restrict__ A, const __bf16* __restrict__ B,
    float* __restrict__ C, int M, int N, int K)
{
    int gw = (blockIdx.x * blockDim.x + threadIdx.x) >> 5;
    int nw = (gridDim.x * blockDim.x) >> 5;
    int lane = threadIdx.x & 31, r = lane & 15, g = lane >> 4;
    int tilesN = N >> 4;
    int tiles  = (M >> 4) * tilesN;
    for (int t = gw; t < tiles; t += nw) {
        int mt = t / tilesN, nt = t - mt * tilesN;
        const __bf16* arow = A + (size_t)(mt * 16 + r) * K;
        const __bf16* brow = B + (size_t)(nt * 16 + r) * K;
        v8f acc = {};
        for (int kb = 0; kb < K; kb += 32) {
            __builtin_prefetch(brow + kb + 64, 0, 0);
            v16bf a, b;
#pragma unroll
            for (int e = 0; e < 16; ++e) {
                int kk = kb + wmma_k(e, g);
                a[e] = arow[kk];
                b[e] = brow[kk];
            }
            acc = __builtin_amdgcn_wmma_f32_16x16x32_bf16(
                false, a, false, b, (short)0, acc, false, false);
        }
        float* crow = C + (size_t)(mt * 16 + g * 8) * N + nt * 16 + r;
#pragma unroll
        for (int i = 0; i < 8; ++i) crow[(size_t)i * N] = acc[i];
    }
}

// ---- GCN graph plumbing -------------------------------------------------
__global__ void k_deg_init(float* deg, int n) {
    for (int i = blockIdx.x * blockDim.x + threadIdx.x; i < n; i += gridDim.x * blockDim.x)
        deg[i] = 1.f;   // self loop
}
__global__ void k_deg_acc(const int* __restrict__ ei, int E, float* deg) {
    for (int e = blockIdx.x * blockDim.x + threadIdx.x; e < E; e += gridDim.x * blockDim.x)
        atomicAdd(&deg[ei[E + e]], 1.f);
}
__global__ void k_dinv(const float* deg, float* dinv, int n) {
    for (int i = blockIdx.x * blockDim.x + threadIdx.x; i < n; i += gridDim.x * blockDim.x)
        dinv[i] = rsqrtf(deg[i]);
}
// one block per edge, 256 cols
__global__ __launch_bounds__(256) void k_scatter_edges(
    const int* __restrict__ ei, int E, const float* __restrict__ dinv,
    const float* __restrict__ H, float* __restrict__ C)
{
    int e = blockIdx.x;
    int s = ei[e], d = ei[E + e];
    float nr = dinv[s] * dinv[d];
    atomicAdd(&C[(size_t)d * 256 + threadIdx.x], nr * H[(size_t)s * 256 + threadIdx.x]);
}
__global__ __launch_bounds__(256) void k_selfloop(
    const float* __restrict__ dinv, const float* __restrict__ H, float* __restrict__ C)
{
    int i = blockIdx.x;
    float nr = dinv[i] * dinv[i];
    C[(size_t)i * 256 + threadIdx.x] += nr * H[(size_t)i * 256 + threadIdx.x];
}
__global__ __launch_bounds__(256) void k_add_bias(float* C, const float* b) {
    C[(size_t)blockIdx.x * 256 + threadIdx.x] += b[threadIdx.x];
}
__global__ __launch_bounds__(256) void k_bias_relu(float* C, const float* b) {
    size_t o = (size_t)blockIdx.x * 256 + threadIdx.x;
    C[o] = fmaxf(C[o] + b[threadIdx.x], 0.f);
}
// c2_in[i] = relu(concat(c1[i], s2v[root]))  -> bf16 [2048][512]
__global__ __launch_bounds__(256) void k_build_c2in(
    const float* __restrict__ c1, const float* __restrict__ s2v,
    const int* __restrict__ root, __bf16* __restrict__ out)
{
    int i = blockIdx.x, c = threadIdx.x;
    int rt = *root;
    out[(size_t)i * 512 + c]       = (__bf16)fmaxf(c1[(size_t)i * 256 + c], 0.f);
    out[(size_t)i * 512 + 256 + c] = (__bf16)fmaxf(s2v[(size_t)rt * 256 + c], 0.f);
}
// branch output: [c1[root] (256) | mean_i c2[i] (256)]
__global__ __launch_bounds__(256) void k_branch_out(
    const float* __restrict__ c1, const float* __restrict__ c2,
    const int* __restrict__ root, float* __restrict__ outp, int n)
{
    int c = threadIdx.x;
    int rt = *root;
    outp[c] = c1[(size_t)rt * 256 + c];
    float s = 0.f;
    for (int i = 0; i < n; ++i) s += c2[(size_t)i * 256 + c];
    outp[256 + c] = s / (float)n;
}
// out[4] = cat[1024] @ fc_w^T + fc_b
__global__ __launch_bounds__(256) void k_fc(
    const float* __restrict__ cat, const float* __restrict__ fcw,
    const float* __restrict__ fcb, float* __restrict__ out)
{
    __shared__ float red[4][64];
    int g = threadIdx.x >> 6, l = threadIdx.x & 63;
    float s = 0.f;
    for (int i = l; i < 1024; i += 64) s += fcw[g * 1024 + i] * cat[i];
    red[g][l] = s;
    __syncthreads();
    if (l == 0) {
        float t = 0.f;
        for (int k = 0; k < 64; ++k) t += red[g][k];
        out[g] = t + fcb[g];
    }
}

extern "C" void kernel_launch(void* const* d_in, const int* in_sizes, int n_in,
                              void* d_out, int out_size, void* d_ws, size_t ws_size,
                              hipStream_t stream) {
    const int*   tokens = (const int*)d_in[0];
    const int*   eiTD   = (const int*)d_in[1];
    const int*   eiBU   = (const int*)d_in[2];
    const int*   root   = (const int*)d_in[3];
    const float* embed  = (const float*)d_in[5];
    const float* inW    = (const float*)d_in[6];
    const float* inB    = (const float*)d_in[7];
    const float* outW   = (const float*)d_in[8];
    const float* outB   = (const float*)d_in[9];
    const float* s2vW   = (const float*)d_in[10];
    const float* s2vB   = (const float*)d_in[11];
    const float* tdw1   = (const float*)d_in[12];
    const float* tdb1   = (const float*)d_in[13];
    const float* tdw2   = (const float*)d_in[14];
    const float* tdb2   = (const float*)d_in[15];
    const float* buw1   = (const float*)d_in[16];
    const float* bub1   = (const float*)d_in[17];
    const float* buw2   = (const float*)d_in[18];
    const float* bub2   = (const float*)d_in[19];
    const float* fcw    = (const float*)d_in[20];
    const float* fcb    = (const float*)d_in[21];
    float* out = (float*)d_out;

    const int N = in_sizes[0] / SEQL;   // 2048 posts
    const int E = in_sizes[1] / 2;      // 2047 edges

    char* ws = (char*)d_ws;
    size_t off = 0;
    auto alloc = [&](size_t b) { size_t o = off; off += (b + 255) & ~(size_t)255; return o; };
    __bf16* wkv_bf = (__bf16*)(ws + alloc((size_t)NKVPAD * KPAD * 2));
    __bf16* w1td   = (__bf16*)(ws + alloc((size_t)256 * 256 * 2));
    __bf16* w1bu   = (__bf16*)(ws + alloc((size_t)256 * 256 * 2));
    __bf16* w2td   = (__bf16*)(ws + alloc((size_t)256 * 512 * 2));
    __bf16* w2bu   = (__bf16*)(ws + alloc((size_t)256 * 512 * 2));
    float*  s2v    = (float*) (ws + alloc((size_t)N * 256 * 4));
    __bf16* s2v_bf = (__bf16*)(ws + alloc((size_t)N * 256 * 2));
    float*  H      = (float*) (ws + alloc((size_t)N * 256 * 4));
    float*  c1     = (float*) (ws + alloc((size_t)N * 256 * 4));
    __bf16* c2in   = (__bf16*)(ws + alloc((size_t)N * 512 * 2));
    float*  c2     = (float*) (ws + alloc((size_t)N * 256 * 4));
    float*  deg    = (float*) (ws + alloc((size_t)N * 4));
    float*  dinv   = (float*) (ws + alloc((size_t)N * 4));
    float*  cat    = (float*) (ws + alloc((size_t)2 * 512 * 4));

    // weight conversion to bf16 (L2-resident afterwards)
    k_cvt_bf16<<<256, 256, 0, stream>>>(inW, wkv_bf, NKVPAD, NKV, KPAD, DMODEL, 300);
    k_cvt_bf16<<<64, 256, 0, stream>>>(tdw1, w1td, 256, 256, 256, 256, 0);
    k_cvt_bf16<<<64, 256, 0, stream>>>(buw1, w1bu, 256, 256, 256, 256, 0);
    k_cvt_bf16<<<128, 256, 0, stream>>>(tdw2, w2td, 256, 256, 512, 512, 0);
    k_cvt_bf16<<<128, 256, 0, stream>>>(buw2, w2bu, 256, 256, 512, 512, 0);

    // fused post encoder (dominant GEMM on WMMA, TDM gather for embeddings)
    k_post_encoder<<<N, 256, 0, stream>>>(tokens, embed, inW, inB, wkv_bf,
                                          outW, outB, s2vW, s2vB, s2v);
    k_f32_to_bf16<<<256, 256, 0, stream>>>(s2v, s2v_bf, N * 256);

    for (int br = 0; br < 2; ++br) {
        const int*    ei = br ? eiBU : eiTD;
        const __bf16* w1 = br ? w1bu : w1td;
        const __bf16* w2 = br ? w2bu : w2td;
        const float*  b1 = br ? bub1 : tdb1;
        const float*  b2 = br ? bub2 : tdb2;
        float* bout = cat + br * 512;

        k_deg_init<<<8, 256, 0, stream>>>(deg, N);
        k_deg_acc<<<(E + 255) / 256, 256, 0, stream>>>(ei, E, deg);
        k_dinv<<<8, 256, 0, stream>>>(deg, dinv, N);

        // conv1: H = s2v @ w1^T ; c1 = scatter(norm*H) + b1
        k_gemm_bf16<<<128, 256, 0, stream>>>(s2v_bf, w1, H, N, 256, 256);
        k_zero<<<256, 256, 0, stream>>>(c1, N * 256);
        k_scatter_edges<<<E, 256, 0, stream>>>(ei, E, dinv, H, c1);
        k_selfloop<<<N, 256, 0, stream>>>(dinv, H, c1);
        k_add_bias<<<N, 256, 0, stream>>>(c1, b1);

        // conv2 input: relu(concat(c1, s2v[root]))
        k_build_c2in<<<N, 256, 0, stream>>>(c1, s2v, root, c2in);
        k_gemm_bf16<<<128, 256, 0, stream>>>(c2in, w2, H, N, 256, 512);
        k_zero<<<256, 256, 0, stream>>>(c2, N * 256);
        k_scatter_edges<<<E, 256, 0, stream>>>(ei, E, dinv, H, c2);
        k_selfloop<<<N, 256, 0, stream>>>(dinv, H, c2);
        k_bias_relu<<<N, 256, 0, stream>>>(c2, b2);

        k_branch_out<<<1, 256, 0, stream>>>(c1, c2, root, bout, N);
    }

    k_fc<<<1, 256, 0, stream>>>(cat, fcw, fcb, out);
}
